// VQClusterHead_55473797595653
// MI455X (gfx1250) — matrix operand order repaired and checked by reading, not compiled, not run on hardware
//
#include <hip/hip_runtime.h>
#include <hip/hip_bf16.h>
#include <stdint.h>

// ---------------------------------------------------------------------------
// VQ + ClusterLookup head for MI455X (gfx1250, wave32, WMMA).
//
//   idx   = argmax_n( x·centers^T + gumbel )        (bf16x3 WMMA GEMM, fused argmax)
//   M     = n(centers)·n(clusters)^T  [1024x1024]   (bf16x4 WMMA GEMM, fp32-class)
//   ip    = M[idx]          (row gather, 134 MB, NT stores)
//   z_q   = centers[idx]    (row gather, 67 MB, NT stores)
//   preds = argmax_j M[idx,j]; loss = -mean(max_j M[idx,j])   (per-code LUTs)
//
// L2 policy: x/centers/M kept regular-temporal (all fit in 192MB L2 -> the
// per-n-chunk x re-reads are L2 hits); gumbel loads and ip/z_q stores are
// non-temporal so the streams do not evict the resident working set.
// ---------------------------------------------------------------------------

typedef __attribute__((ext_vector_type(16))) __bf16       bf16x16;
typedef __attribute__((ext_vector_type(8)))  float        f32x8;
typedef __attribute__((ext_vector_type(4)))  unsigned int u32x4;

union FragAB { bf16x16 v; u32x4 q[2]; };

#define WMMA_BF16(A, B, C) \
  __builtin_amdgcn_wmma_f32_16x16x32_bf16(false, (A), false, (B), (short)0, (C), false, false)

#define B_   8
#define C_   512
#define HW_  4096
#define N_   1024

// ---------------------------------------------------------------------- K0: row norms
__global__ __launch_bounds__(256) void k_norms(const float* __restrict__ centers,
                                               const float* __restrict__ clusters,
                                               float* __restrict__ invC,
                                               float* __restrict__ invCl) {
  const int row = blockIdx.x;                     // 0..2047
  const float* src = (row < N_) ? centers + (size_t)row * C_
                                : clusters + (size_t)(row - N_) * C_;
  float s = 0.f;
  for (int i = threadIdx.x; i < C_; i += 256) { float v = src[i]; s += v * v; }
  for (int off = 16; off; off >>= 1) s += __shfl_down(s, off, 32);
  __shared__ float red[8];
  if ((threadIdx.x & 31) == 0) red[threadIdx.x >> 5] = s;
  __syncthreads();
  if (threadIdx.x == 0) {
    float t = 0.f;
    for (int i = 0; i < 8; ++i) t += red[i];
    float inv = 1.f / fmaxf(sqrtf(t), 1e-12f);    // F.normalize eps
    if (row < N_) invC[row] = inv; else invCl[row - N_] = inv;
  }
}

// ------------------------------------------------------- K0b: centers -> bf16 hi/lo
__global__ __launch_bounds__(256) void k_split(const float* __restrict__ centers,
                                               __bf16* __restrict__ hi,
                                               __bf16* __restrict__ lo) {
  size_t i = (size_t)blockIdx.x * 256 + threadIdx.x;   // grid = 2048 -> 512K elems
  float v = centers[i];
  __bf16 h = (__bf16)v;
  hi[i] = h;
  lo[i] = (__bf16)(v - (float)h);
}

// ------------------------------------- K1: M = n(centers) x n(clusters)^T  (bf16x4)
__global__ __launch_bounds__(256) void k_gemmM(const float* __restrict__ centers,
                                               const float* __restrict__ clusters,
                                               const float* __restrict__ invC,
                                               const float* __restrict__ invCl,
                                               float* __restrict__ M) {
  __shared__ __bf16 Ah[128 * 40], Al[128 * 40];   // [n][k], padded to 40 (bank stagger)
  __shared__ __bf16 Bh[32 * 72],  Bl[32 * 72];    // [n][k], stride 32, unpadded
  const int tid = threadIdx.x, wv = tid >> 5, ln = tid & 31;
  const int rowbase = (blockIdx.x >> 4) * 128;    // 8 row tiles
  const int colbase = (blockIdx.x & 15) * 64;     // 16 col tiles

  f32x8 acc[4];
  for (int t = 0; t < 4; ++t) for (int e = 0; e < 8; ++e) acc[t][e] = 0.f;

  for (int ks = 0; ks < 16; ++ks) {               // K in steps of 32
    for (int i = 0; i < 16; ++i) {                // A tile: 128x32 normalized centers
      int idx = i * 256 + tid, n = idx >> 5, k = idx & 31;
      float v = centers[(size_t)(rowbase + n) * C_ + ks * 32 + k] * invC[rowbase + n];
      __bf16 h = (__bf16)v;
      Ah[n * 40 + k] = h; Al[n * 40 + k] = (__bf16)(v - (float)h);
    }
    for (int i = 0; i < 8; ++i) {                 // B tile: [n][k], stride 32
      int idx = i * 256 + tid, n = idx >> 5, k = idx & 31;
      float v = clusters[(size_t)(colbase + n) * C_ + ks * 32 + k] * invCl[colbase + n];
      __bf16 h = (__bf16)v;
      Bh[n * 32 + k] = h; Bl[n * 32 + k] = (__bf16)(v - (float)h);
    }
    __syncthreads();

    // A fragment: lane&15 = row, lanes<16 -> K {0..7,16..23}, lanes>=16 -> {8..15,24..31}
    const int r = wv * 16 + (ln & 15);
    const int kk = (ln < 16) ? 0 : 8;
    FragAB fah, fal;
    fah.q[0] = *(const u32x4*)&Ah[r * 40 + kk]; fah.q[1] = *(const u32x4*)&Ah[r * 40 + kk + 16];
    fal.q[0] = *(const u32x4*)&Al[r * 40 + kk]; fal.q[1] = *(const u32x4*)&Al[r * 40 + kk + 16];

    const int kr = (ln & 15) + ((ln < 16) ? 0 : 16);   // kr unused; B: lane&15 = N col, K split as A
    for (int cb = 0; cb < 4; ++cb) {
      FragAB fbh, fbl;
      fbh.q[0] = *(const u32x4*)&Bh[(cb * 16 + (ln & 15)) * 32 + kk];
      fbh.q[1] = *(const u32x4*)&Bh[(cb * 16 + (ln & 15)) * 32 + kk + 16];
      fbl.q[0] = *(const u32x4*)&Bl[(cb * 16 + (ln & 15)) * 32 + kk];
      fbl.q[1] = *(const u32x4*)&Bl[(cb * 16 + (ln & 15)) * 32 + kk + 16];
      acc[cb] = WMMA_BF16(fah.v, fbh.v, acc[cb]);
      acc[cb] = WMMA_BF16(fah.v, fbl.v, acc[cb]);
      acc[cb] = WMMA_BF16(fal.v, fbh.v, acc[cb]);
      acc[cb] = WMMA_BF16(fal.v, fbl.v, acc[cb]);
    }
    __syncthreads();
  }
  for (int cb = 0; cb < 4; ++cb)
    for (int r2 = 0; r2 < 8; ++r2) {
      int mr = rowbase + wv * 16 + r2 + ((ln < 16) ? 0 : 8);
      int mc = colbase + cb * 16 + (ln & 15);
      M[(size_t)mr * N_ + mc] = acc[cb][r2];
    }
}

// ---------------------------------------- K2: per-code argmax/max LUTs over M rows
__global__ __launch_bounds__(256) void k_rowmax(const float* __restrict__ M,
                                                int* __restrict__ predOf,
                                                float* __restrict__ maxOf) {
  const int row = blockIdx.x;
  const float* mr = M + (size_t)row * N_;
  float bv = -3.4e38f; int bj = 0x7fffffff;
  for (int j = threadIdx.x; j < N_; j += 256) {
    float v = mr[j];
    if (v > bv || (v == bv && j < bj)) { bv = v; bj = j; }
  }
  __shared__ float rv[256]; __shared__ int rj[256];
  rv[threadIdx.x] = bv; rj[threadIdx.x] = bj;
  __syncthreads();
  for (int off = 128; off; off >>= 1) {
    if (threadIdx.x < off) {
      float v2 = rv[threadIdx.x + off]; int j2 = rj[threadIdx.x + off];
      if (v2 > rv[threadIdx.x] || (v2 == rv[threadIdx.x] && j2 < rj[threadIdx.x])) {
        rv[threadIdx.x] = v2; rj[threadIdx.x] = j2;
      }
    }
    __syncthreads();
  }
  if (threadIdx.x == 0) { predOf[row] = rj[0]; maxOf[row] = rv[0]; }
}

// --------------- K3: fused GEMM(+gumbel) argmax -> idx -> scatter ip/z_q/preds/loss
// One workgroup (8 waves) owns 64 contiguous pixels of one batch image; n streamed
// in 4 chunks of 256 with running (max, argmax) in registers. bf16x3 split GEMM.
// Ping-pong LDS x tile: 1 barrier per k-step, global loads overlap WMMA chain.
__global__ __launch_bounds__(256) void k_main(const float* __restrict__ x,
                                              const float* __restrict__ gumbel,
                                              const __bf16* __restrict__ centHi,
                                              const __bf16* __restrict__ centLo,
                                              const float* __restrict__ centers,
                                              const float* __restrict__ M,
                                              const int* __restrict__ predOf,
                                              const float* __restrict__ maxOf,
                                              float* __restrict__ partials,
                                              float* __restrict__ preds_out,
                                              float* __restrict__ ip_out,
                                              float* __restrict__ zq_out) {
  __shared__ __bf16 Xh[2][32 * 72], Xl[2][32 * 72];  // ping-pong x tiles [p][c]
  __shared__ float redV[512]; __shared__ int redN[512];
  __shared__ int idxArr[64]; __shared__ float lred[64];

  const int tid = threadIdx.x, wv = tid >> 5, ln = tid & 31;
  const int b = blockIdx.x >> 6;
  const int hw0 = (blockIdx.x & 63) * 64;
  const float* xb = x + (size_t)b * C_ * HW_ + hw0;

  const int s_c = tid >> 6, s_p = tid & 63;       // staging coords (4 rows apart)

  float stage[8];
  auto loadStage = [&](int ks) {
#pragma unroll
    for (int i = 0; i < 8; ++i)
      stage[i] = xb[(size_t)(ks * 32 + s_c + i * 4) * HW_ + s_p];
  };
  auto storeStage = [&](int buf) {
#pragma unroll
    for (int i = 0; i < 8; ++i) {
      float v = stage[i];
      __bf16 h = (__bf16)v;
      Xh[buf][s_p * 32 + s_c + i * 4] = h;
      Xl[buf][s_p * 32 + s_c + i * 4] = (__bf16)(v - (float)h);
    }
  };

  float bestV[4]; int bestN[4];
  for (int t = 0; t < 4; ++t) { bestV[t] = -3.4e38f; bestN[t] = 0x7fffffff; }

  for (int nc = 0; nc < 4; ++nc) {                // n chunk of 256
    const int n0 = nc * 256;

    // prefetch this chunk's gumbel block (256 rows x 64 px): 2 lines per thread
    {
      const float* pf = gumbel + ((size_t)b * N_ + n0 + tid) * HW_ + hw0;
      __builtin_prefetch(pf, 0, 0);
      __builtin_prefetch(pf + 32, 0, 0);
    }

    f32x8 acc[2][4];
    for (int a = 0; a < 2; ++a)
      for (int t = 0; t < 4; ++t)
        for (int e = 0; e < 8; ++e) acc[a][t][e] = 0.f;

    loadStage(0); storeStage(0);
    __syncthreads();

    for (int ks = 0; ks < 16; ++ks) {             // C in steps of 32
      const int cur = ks & 1;
      if (ks < 15) loadStage(ks + 1);             // overlap with WMMAs below

      const int kr = (ln & 15) + ((ln < 16) ? 0 : 16);
      const int kk = (ln < 16) ? 0 : 8;
      for (int rbi = 0; rbi < 2; ++rbi) {
        const int n_g = n0 + (wv + rbi * 8) * 16 + (ln & 15);
        const __bf16* ah = centHi + (size_t)n_g * C_ + ks * 32 + kk;
        const __bf16* al = centLo + (size_t)n_g * C_ + ks * 32 + kk;
        FragAB fah, fal;
        fah.q[0] = *(const u32x4*)ah; fah.q[1] = *(const u32x4*)(ah + 16);
        fal.q[0] = *(const u32x4*)al; fal.q[1] = *(const u32x4*)(al + 16);
        for (int t = 0; t < 4; ++t) {
          FragAB fbh, fbl;
          fbh.q[0] = *(const u32x4*)&Xh[cur][(t * 16 + (ln & 15)) * 32 + kk];
          fbh.q[1] = *(const u32x4*)&Xh[cur][(t * 16 + (ln & 15)) * 32 + kk + 16];
          fbl.q[0] = *(const u32x4*)&Xl[cur][(t * 16 + (ln & 15)) * 32 + kk];
          fbl.q[1] = *(const u32x4*)&Xl[cur][(t * 16 + (ln & 15)) * 32 + kk + 16];
          acc[rbi][t] = WMMA_BF16(fah.v, fbh.v, acc[rbi][t]);   // hi*hi
          acc[rbi][t] = WMMA_BF16(fah.v, fbl.v, acc[rbi][t]);   // hi*lo
          acc[rbi][t] = WMMA_BF16(fal.v, fbh.v, acc[rbi][t]);   // lo*hi
        }
      }
      if (ks < 15) storeStage(cur ^ 1);
      __syncthreads();
    }

    // add gumbel (non-temporal stream), update running per-pixel (max, argmax)
    for (int t = 0; t < 4; ++t) {
      const int p = hw0 + t * 16 + (ln & 15);
      for (int rbi = 0; rbi < 2; ++rbi) {
        const int nb = n0 + (wv + rbi * 8) * 16 + ((ln < 16) ? 0 : 8);
        const float* g = gumbel + ((size_t)b * N_ + nb) * HW_ + p;
        for (int r = 0; r < 8; ++r) {
          float v = acc[rbi][t][r] + __builtin_nontemporal_load(&g[(size_t)r * HW_]);
          int n = nb + r;
          if (v > bestV[t] || (v == bestV[t] && n < bestN[t])) { bestV[t] = v; bestN[t] = n; }
        }
      }
    }
  }

  // combine lane L with L^16 (same pixel, different n half)
  for (int t = 0; t < 4; ++t) {
    float ov = __shfl_xor(bestV[t], 16, 32);
    int   on = __shfl_xor(bestN[t], 16, 32);
    if (ov > bestV[t] || (ov == bestV[t] && on < bestN[t])) { bestV[t] = ov; bestN[t] = on; }
  }
  if (ln < 16)
    for (int t = 0; t < 4; ++t) {
      redV[wv * 64 + t * 16 + ln] = bestV[t];
      redN[wv * 64 + t * 16 + ln] = bestN[t];
    }
  __syncthreads();

  if (tid < 64) {                                 // cross-wave reduce (fixed order)
    float bv = redV[tid]; int bn = redN[tid];
    for (int w2 = 1; w2 < 8; ++w2) {
      float v2 = redV[w2 * 64 + tid]; int n2 = redN[w2 * 64 + tid];
      if (v2 > bv || (v2 == bv && n2 < bn)) { bv = v2; bn = n2; }
    }
    idxArr[tid] = bn;
    preds_out[(size_t)b * HW_ + hw0 + tid] = (float)predOf[bn];
    lred[tid] = maxOf[bn];
  }
  __syncthreads();
  if (tid == 0) {
    float s = 0.f;
    for (int i = 0; i < 64; ++i) s += lred[i];    // deterministic partial loss
    partials[blockIdx.x] = s;
  }

  // z_q = centers[idx]  (centers L2-resident; NT stores, 128B-coalesced per wave)
  {
    const int p = tid & 63, cs = tid >> 6;
    const float* crow = centers + (size_t)idxArr[p] * C_;
    float* zq = zq_out + (size_t)b * C_ * HW_ + hw0 + p;
    for (int c = cs; c < C_; c += 4)
      __builtin_nontemporal_store(crow[c], &zq[(size_t)c * HW_]);
  }
  // ip = M[idx]  (M 4MB, L2-resident; NT stores, 128B-coalesced per wave)
  {
    const int p = tid & 63, ns = tid >> 6;
    const float* mrow = M + (size_t)idxArr[p] * N_;
    float* ip = ip_out + (size_t)b * N_ * HW_ + hw0 + p;
    for (int n = ns; n < N_; n += 4)
      __builtin_nontemporal_store(mrow[n], &ip[(size_t)n * HW_]);
  }
}

// ----------------------------------------------------------- K4: final loss reduce
__global__ __launch_bounds__(256) void k_loss(const float* __restrict__ partials,
                                              float* __restrict__ out) {
  __shared__ float red[256];
  red[threadIdx.x] = partials[threadIdx.x] + partials[threadIdx.x + 256];
  __syncthreads();
  for (int off = 128; off; off >>= 1) {
    if (threadIdx.x < off) red[threadIdx.x] += red[threadIdx.x + off];
    __syncthreads();
  }
  if (threadIdx.x == 0) out[0] = -red[0] / 32768.0f;
}

// ---------------------------------------------------------------------------
extern "C" void kernel_launch(void* const* d_in, const int* in_sizes, int n_in,
                              void* d_out, int out_size, void* d_ws, size_t ws_size,
                              hipStream_t stream) {
  const float* x        = (const float*)d_in[0];
  const float* centers  = (const float*)d_in[1];
  const float* clusters = (const float*)d_in[2];
  const float* gumbel   = (const float*)d_in[3];
  float* out = (float*)d_out;

  // workspace layout (~6.31 MB)
  float*  Mmat     = (float*)d_ws;                 // 1024*1024 f32
  float*  invC     = Mmat + 1024 * 1024;           // 1024
  float*  invCl    = invC + 1024;                  // 1024
  int*    predOf   = (int*)(invCl + 1024);         // 1024
  float*  maxOf    = (float*)(predOf + 1024);      // 1024
  float*  partials = maxOf + 1024;                 // 512
  __bf16* centHi   = (__bf16*)(partials + 512);    // 1024*512 bf16 (16B aligned)
  __bf16* centLo   = centHi + 1024 * 512;

  // output layout: [loss(1)][preds(32768)][ip(33554432)][zq(16777216)]
  float* lossOut  = out;
  float* predsOut = out + 1;
  float* ipOut    = predsOut + 32768;
  float* zqOut    = ipOut + (size_t)B_ * N_ * HW_;

  k_norms <<<2048, 256, 0, stream>>>(centers, clusters, invC, invCl);
  k_split <<<2048, 256, 0, stream>>>(centers, centHi, centLo);
  k_gemmM <<<128,  256, 0, stream>>>(centers, clusters, invC, invCl, Mmat);
  k_rowmax<<<1024, 256, 0, stream>>>(Mmat, predOf, maxOf);
  k_main  <<<512,  256, 0, stream>>>(x, gumbel, centHi, centLo, centers, Mmat,
                                     predOf, maxOf, partials, predsOut, ipOut, zqOut);
  k_loss  <<<1,    256, 0, stream>>>(partials, lossOut);
}
